// HybridUnit_8392366096692
// MI455X (gfx1250) — matrix operand order repaired and checked by reading, not compiled, not run on hardware
//
#include <hip/hip_runtime.h>

// Problem constants (from the reference)
#define BB 32
#define TT 512
#define DD 256
#define UU 1024
#define MBIG (BB * TT)   // 16384

typedef __attribute__((ext_vector_type(16))) __bf16 v16bf;
typedef __attribute__((ext_vector_type(8)))  __bf16 v8bf;
typedef __attribute__((ext_vector_type(8)))  float  v8f;

typedef __attribute__((ext_vector_type(4))) unsigned int v4ui;
typedef __attribute__((ext_vector_type(8))) int          v8si;
typedef __attribute__((ext_vector_type(4))) int          v4si;

// ---- fp32 -> bf16 (round-to-nearest-even, manual bit ops) ----
__device__ __forceinline__ unsigned short f32_to_bf16_rne(float f) {
    unsigned int u = __builtin_bit_cast(unsigned int, f);
    unsigned int r = u + 0x7FFFu + ((u >> 16) & 1u);
    return (unsigned short)(r >> 16);
}
__device__ __forceinline__ float bf16_to_f32(unsigned short h) {
    unsigned int u = ((unsigned int)h) << 16;
    return __builtin_bit_cast(float, u);
}
__device__ __forceinline__ void split2(float x, unsigned short &hs, unsigned short &ls) {
    hs = f32_to_bf16_rne(x);
    ls = f32_to_bf16_rne(x - bf16_to_f32(hs));
}

__device__ __forceinline__ float softsign(float x) {
    return x / (1.0f + fabsf(x));
}

// ============================================================================
// One-time split pass: fp32 [n4*4] -> bf16 hi/lo arrays (same layout).
// ============================================================================
__global__ __launch_bounds__(256)
void split_f32_bf16(const float* __restrict__ in,
                    unsigned short* __restrict__ hi,
                    unsigned short* __restrict__ lo, int n4)
{
    int i = blockIdx.x * blockDim.x + threadIdx.x;
    if (i >= n4) return;
    float4 x = ((const float4*)in)[i];
    unsigned short h0, l0, h1, l1, h2, l2, h3, l3;
    split2(x.x, h0, l0); split2(x.y, h1, l1);
    split2(x.z, h2, l2); split2(x.w, h3, l3);
    uint2 hp, lp;
    hp.x = (unsigned)h0 | ((unsigned)h1 << 16);
    hp.y = (unsigned)h2 | ((unsigned)h3 << 16);
    lp.x = (unsigned)l0 | ((unsigned)l1 << 16);
    lp.y = (unsigned)l2 | ((unsigned)l3 << 16);
    ((uint2*)hi)[i] = hp;
    ((uint2*)lo)[i] = lp;
}

// ============================================================================
// One-time split + transpose: W [256,1024] fp32 -> Wt hi/lo [1024,256] bf16.
// ============================================================================
__global__ __launch_bounds__(256)
void split_transpose(const float* __restrict__ W,
                     unsigned short* __restrict__ hi,
                     unsigned short* __restrict__ lo)
{
    int id = blockIdx.x * 256 + threadIdx.x;   // 0 .. DD*UU-1, id = n*DD + k
    int k = id & (DD - 1);
    int n = id >> 8;                           // DD == 256
    float x = W[(size_t)k * UU + n];
    unsigned short hs, ls;
    split2(x, hs, ls);
    hi[id] = hs;
    lo[id] = ls;
}

// ============================================================================
// C[M,1024] = A[M,256] @ W[256,1024], operands pre-split to bf16 hi/lo.
//   A hi/lo: row-major [M,256] bf16.  Wt hi/lo: transposed [1024,256] bf16.
// Block = 4 waves; block tile = 64 rows x 64 cols; wave owns a 16-col strip.
// The 64x256 weight tile (hi+lo, 32KB each) is staged into LDS once per
// block by the Tensor Data Mover, then reused for 4 M-tiles.
// K loop: 8 chunks of 32, 3 v_wmma_f32_16x16x32_bf16 per chunk (bf16x3).
// ============================================================================
__global__ __launch_bounds__(128)
void wmma_gemm_tdm(const unsigned short* __restrict__ Ahi_,
                   const unsigned short* __restrict__ Alo_,
                   const unsigned short* __restrict__ Wthi_,
                   const unsigned short* __restrict__ Wtlo_,
                   float* __restrict__ C,
                   int M)
{
    __shared__ unsigned short smem_hi[64 * DD];   // 32 KB
    __shared__ unsigned short smem_lo[64 * DD];   // 32 KB

    const int wave  = threadIdx.x >> 5;
    const int lane  = threadIdx.x & 31;
    const int lrow  = lane & 15;
    const int lhalf = lane >> 4;            // 0 = lanes 0-15, 1 = lanes 16-31

    const int n0 = blockIdx.y * 64;         // block's first output column

    // ---- TDM: stage Wt[n0 .. n0+63][0..255] hi & lo tiles into LDS --------
    // Transposed layout makes the tile one contiguous 32KB block: describe it
    // as a 1D tensor of 4096 8-byte elements (D# per ISA 8.3/8.4).
    if (wave == 0) {
        unsigned long long gh = (unsigned long long)(uintptr_t)(Wthi_ + (size_t)n0 * DD);
        unsigned long long gl = (unsigned long long)(uintptr_t)(Wtlo_ + (size_t)n0 * DD);
        unsigned ldsh = (unsigned)(uintptr_t)&smem_hi[0];
        unsigned ldsl = (unsigned)(uintptr_t)&smem_lo[0];

        // group1: data_size=3 (8B); tensor_dim0=4096; tensor_dim1=1;
        //         tile_dim0=4096; tensor_dim0_stride=4096
        v8si g1 = { 0x00030000, 0x10000000, 0x00010000, 0x10000000,
                    0, 4096, 0, 0 };
        v4si gz4 = { 0, 0, 0, 0 };
        v8si gz8 = { 0, 0, 0, 0, 0, 0, 0, 0 };

        v4ui g0h = { 1u, ldsh, (unsigned)gh,
                     (unsigned)((gh >> 32) & 0x1FFFFFFu) | (2u << 30) };
        __builtin_amdgcn_tensor_load_to_lds(g0h, g1, gz4, gz4, gz8, 0);

        v4ui g0l = { 1u, ldsl, (unsigned)gl,
                     (unsigned)((gl >> 32) & 0x1FFFFFFu) | (2u << 30) };
        __builtin_amdgcn_tensor_load_to_lds(g0l, g1, gz4, gz4, gz8, 0);

        __builtin_amdgcn_s_wait_tensorcnt(0);
    }
    __syncthreads();

    const __bf16* Ahi = (const __bf16*)Ahi_;
    const __bf16* Alo = (const __bf16*)Alo_;
    // This wave's B column within the LDS tile:
    const int bcol = wave * 16 + lrow;               // 0..63
    const __bf16* Bh = (const __bf16*)&smem_hi[bcol * DD];
    const __bf16* Bl = (const __bf16*)&smem_lo[bcol * DD];

    // ---- 4 M-tiles per block, reusing the staged weight tile --------------
    for (int mt = 0; mt < 4; ++mt) {
        const int m0 = blockIdx.x * 64 + mt * 16;
        if (m0 >= M) break;

        const __bf16* ArowH = Ahi + (size_t)(m0 + lrow) * DD;
        const __bf16* ArowL = Alo + (size_t)(m0 + lrow) * DD;

        v8f acc = {};
        #pragma unroll
        for (int k0 = 0; k0 < DD; k0 += 32) {
            // A fragment (16x32 bf16, contiguous k = 16*lhalf + s order): lane-half picks K offset.
            const int ka = k0 + lhalf * 16;
            v8bf a0h = *(const v8bf*)(ArowH + ka);
            v8bf a1h = *(const v8bf*)(ArowH + ka + 8);
            v8bf a0l = *(const v8bf*)(ArowL + ka);
            v8bf a1l = *(const v8bf*)(ArowL + ka + 8);
            v16bf ah = __builtin_shufflevector(a0h, a1h,
                0,1,2,3,4,5,6,7,8,9,10,11,12,13,14,15);
            v16bf al = __builtin_shufflevector(a0l, a1l,
                0,1,2,3,4,5,6,7,8,9,10,11,12,13,14,15);
            // B fragment (32x16 bf16, same contiguous k order as A): K-run is contiguous in LDS.
            const int kb = k0 + lhalf * 16;
            v16bf bh = *(const v16bf*)(Bh + kb);
            v16bf bl = *(const v16bf*)(Bl + kb);

            // (Ah+Al)(Bh+Bl) ~= AhBh + AlBh + AhBl  (bf16x3 fp32 emulation)
            acc = __builtin_amdgcn_wmma_f32_16x16x32_bf16(false, ah, false, bh,
                                                          (short)0, acc, false, false);
            acc = __builtin_amdgcn_wmma_f32_16x16x32_bf16(false, al, false, bh,
                                                          (short)0, acc, false, false);
            acc = __builtin_amdgcn_wmma_f32_16x16x32_bf16(false, ah, false, bl,
                                                          (short)0, acc, false, false);
        }

        // f32 C/D layout: VGPR v -> row m0 + v + 8*lhalf, col n0 + wave*16 + lrow
        #pragma unroll
        for (int v = 0; v < 8; ++v) {
            C[(size_t)(m0 + v + 8 * lhalf) * UU + (n0 + wave * 16 + lrow)] = acc[v];
        }
    }
}

// ============================================================================
// In-place recurrent scan over d_out (which holds XW after the big GEMM).
// ============================================================================
__global__ __launch_bounds__(256)
void scan_inplace(const float* __restrict__ G0,   // [32,1024] = X_vecs @ A_0
                  const float* __restrict__ Gs,   // [32,1024] = X_vecs @ A_s
                  const float* __restrict__ b0,
                  const float* __restrict__ bs,
                  const float* __restrict__ bd,
                  float* __restrict__ out)        // [32,512,1024], holds XW
{
    const int idx = blockIdx.x * blockDim.x + threadIdx.x;   // 0 .. B*U/4-1
    const int b   = idx / (UU / 4);
    const int u4  = (idx % (UU / 4)) * 4;

    const float4 g0  = *(const float4*)(G0 + (size_t)b * UU + u4);
    const float4 gs  = *(const float4*)(Gs + (size_t)b * UU + u4);
    const float4 vb0 = *(const float4*)(b0 + u4);
    const float4 vbs = *(const float4*)(bs + u4);
    const float4 vbd = *(const float4*)(bd + u4);

    float h0 = softsign(g0.x + vb0.x);
    float h1 = softsign(g0.y + vb0.y);
    float h2 = softsign(g0.z + vb0.z);
    float h3 = softsign(g0.w + vb0.w);

    const float d0 = gs.x + vbs.x + vbd.x;
    const float d1 = gs.y + vbs.y + vbd.y;
    const float d2 = gs.z + vbs.z + vbd.z;
    const float d3 = gs.w + vbs.w + vbd.w;

    float* p = out + (size_t)b * TT * UU + u4;
    float4 x = *(const float4*)p;
    for (int t = 0; t < TT; ++t) {
        float4 xn;
        if (t + 1 < TT) xn = *(const float4*)(p + UU);   // prefetch next step
        h0 = softsign(h0 + d0 + x.x);
        h1 = softsign(h1 + d1 + x.y);
        h2 = softsign(h2 + d2 + x.z);
        h3 = softsign(h3 + d3 + x.w);
        *(float4*)p = make_float4(h0, h1, h2, h3);
        p += UU;
        x = xn;
    }
}

// ============================================================================
extern "C" void kernel_launch(void* const* d_in, const int* in_sizes, int n_in,
                              void* d_out, int out_size, void* d_ws, size_t ws_size,
                              hipStream_t stream) {
    const float* X_vecs = (const float*)d_in[0];   // [32, 256]
    const float* X_ts   = (const float*)d_in[1];   // [32, 512, 256]
    const float* A0     = (const float*)d_in[2];   // [256, 1024]
    const float* b0     = (const float*)d_in[3];   // [1024]
    const float* As     = (const float*)d_in[4];   // [256, 1024]
    const float* bs     = (const float*)d_in[5];   // [1024]
    const float* Wd     = (const float*)d_in[6];   // [256, 1024]
    const float* bd     = (const float*)d_in[7];   // [1024]
    float* out = (float*)d_out;                    // [32, 512, 1024]

    // ---- workspace layout (bf16 operands + small fp32 GEMM results) ----
    unsigned short* us = (unsigned short*)d_ws;
    unsigned short* Ahi  = us;                    us += (size_t)MBIG * DD;
    unsigned short* Alo  = us;                    us += (size_t)MBIG * DD;
    unsigned short* Vhi  = us;                    us += (size_t)BB * DD;
    unsigned short* Vlo  = us;                    us += (size_t)BB * DD;
    unsigned short* Wdhi = us;                    us += (size_t)UU * DD;
    unsigned short* Wdlo = us;                    us += (size_t)UU * DD;
    unsigned short* A0hi = us;                    us += (size_t)UU * DD;
    unsigned short* A0lo = us;                    us += (size_t)UU * DD;
    unsigned short* Ashi = us;                    us += (size_t)UU * DD;
    unsigned short* Aslo = us;                    us += (size_t)UU * DD;
    float* G0f = (float*)us;
    float* Gsf = G0f + BB * UU;

    // ---- one-time bf16 hi/lo conversions (+ weight transposes) ----
    split_f32_bf16<<<(MBIG * DD / 4 + 255) / 256, 256, 0, stream>>>(
        X_ts, Ahi, Alo, MBIG * DD / 4);
    split_f32_bf16<<<(BB * DD / 4 + 255) / 256, 256, 0, stream>>>(
        X_vecs, Vhi, Vlo, BB * DD / 4);
    split_transpose<<<DD * UU / 256, 256, 0, stream>>>(Wd, Wdhi, Wdlo);
    split_transpose<<<DD * UU / 256, 256, 0, stream>>>(A0, A0hi, A0lo);
    split_transpose<<<DD * UU / 256, 256, 0, stream>>>(As, Ashi, Aslo);

    // ---- GEMMs (TDM-staged weights, bf16x3 WMMA) ----
    wmma_gemm_tdm<<<dim3(MBIG / 64, UU / 64), 128, 0, stream>>>(
        Ahi, Alo, Wdhi, Wdlo, out, MBIG);                 // XW -> d_out
    wmma_gemm_tdm<<<dim3(1, UU / 64), 128, 0, stream>>>(
        Vhi, Vlo, A0hi, A0lo, G0f, BB);                   // X_vecs @ A_0
    wmma_gemm_tdm<<<dim3(1, UU / 64), 128, 0, stream>>>(
        Vhi, Vlo, Ashi, Aslo, Gsf, BB);                   // X_vecs @ A_s

    // ---- sequential recurrence, in place over d_out ----
    scan_inplace<<<(BB * UU / 4) / 256, 256, 0, stream>>>(
        G0f, Gsf, b0, bs, bd, out);
}
